// TemporalGraphAttention_12756052869693
// MI455X (gfx1250) — matrix operand order repaired and checked by reading, not compiled, not run on hardware
//
#include <hip/hip_runtime.h>
#include <hip/hip_bf16.h>
#include <stdint.h>

// ---------------- types ----------------
typedef __attribute__((ext_vector_type(16))) __bf16 v16bf;
typedef __attribute__((ext_vector_type(8)))  float  v8f;

#define BB   8
#define SS   2048
#define HID  256
#define NH   8
#define DD   32
#define EE   32768
#define NN   (BB*SS)          // 16384 nodes
#define BE   (BB*EE)          // 262144 batched edges
#define SCALE 0.17677669529663687f

__device__ __forceinline__ unsigned short f2bf(float f) {
  unsigned u = __float_as_uint(f);
  u += 0x7FFFu + ((u >> 16) & 1u);       // round-to-nearest-even
  return (unsigned short)(u >> 16);
}

// ---------------- elementwise helpers ----------------
__global__ void k_f32_to_bf16(const float* __restrict__ in,
                              unsigned short* __restrict__ out, int n) {
  int i = blockIdx.x * 256 + threadIdx.x;
  if (i < n) out[i] = f2bf(in[i]);
}

__global__ void k_zero_f32(float* __restrict__ p, int n) {
  int i = blockIdx.x * 256 + threadIdx.x;
  if (i < n) p[i] = 0.f;
}

// ---------------- WMMA GEMM: C[M,N] = A[M,K](bf16) * W[K,N](bf16) + bias ----
// Workgroup: 256 threads = 8 waves. Tile: 128 (M) x 64 (N), K-step 32.
// Double-buffered LDS; A tile staged via GLOBAL_LOAD_ASYNC_TO_LDS_B128
// (ASYNCcnt-tracked DMA), B tile staged transposed so fragments are
// contiguous ds_load_b128 pairs.
#define TM 128
#define TN 64
#define KT 32

__global__ __launch_bounds__(256)
void k_gemm_bf16_wmma(const unsigned short* __restrict__ A,
                      const unsigned short* __restrict__ W,
                      const float* __restrict__ bias,
                      float* __restrict__ C,
                      int M, int N, int K)
{
  __shared__ unsigned short As[2][TM * KT];   // 2 x 8 KB, row-major [m][k]
  __shared__ unsigned short Bs[2][TN * KT];   // 2 x 4 KB, transposed [n][k]

  const int tid  = threadIdx.x;
  const int lane = tid & 31;
  const int wv   = tid >> 5;      // wave 0..7
  const int ln   = lane & 15;
  const int half = lane >> 4;     // 0 or 1
  const int rowbase = blockIdx.y * TM;
  const int colbase = blockIdx.x * TN;

  // staging assignments
  const int ar = tid >> 1;            // 0..127  (A row within tile)
  const int ac = (tid & 1) * 16;      // 0 or 16 (A col chunk)
  const int bk = tid & 31;            // 0..31   (B row k within tile)
  const int bn = (tid >> 5) * 8;      // 0..56   (B col group)

  const unsigned short* Agp = A + (size_t)(rowbase + ar) * K + ac;  // + k0
  const unsigned short* Wgp = W + (size_t)bk * N + colbase + bn;    // + k0*N

  v8f zero = {0.f,0.f,0.f,0.f,0.f,0.f,0.f,0.f};
  v8f acc[4];
  acc[0] = zero; acc[1] = zero; acc[2] = zero; acc[3] = zero;

  // ---- prologue: stage k0 = 0 into buffer 0 ----
  {
    unsigned lo = (unsigned)(uintptr_t)(&As[0][ar * KT + ac]);
    unsigned long long ga = (unsigned long long)(uintptr_t)Agp;
    asm volatile("global_load_async_to_lds_b128 %0, %1, off\n\t"
                 "global_load_async_to_lds_b128 %0, %1, off offset:16"
                 :: "v"(lo), "v"(ga) : "memory");
    union { uint4 q; unsigned short s[8]; } bu;
    bu.q = *reinterpret_cast<const uint4*>(Wgp);
#pragma unroll
    for (int ii = 0; ii < 8; ++ii)
      Bs[0][(bn + ii) * KT + bk] = bu.s[ii];
    asm volatile("s_wait_asynccnt 0x0" ::: "memory");
  }
  __syncthreads();

  int buf = 0;
  for (int k0 = 0; k0 < K; k0 += KT) {
    const int nk = k0 + KT;
    union { uint4 q; unsigned short s[8]; } bun;
    if (nk < K) {
      // prefetch next tile: async A DMA into LDS, B row into registers
      unsigned lo = (unsigned)(uintptr_t)(&As[buf ^ 1][ar * KT + ac]);
      unsigned long long ga = (unsigned long long)(uintptr_t)(Agp + nk);
      asm volatile("global_load_async_to_lds_b128 %0, %1, off\n\t"
                   "global_load_async_to_lds_b128 %0, %1, off offset:16"
                   :: "v"(lo), "v"(ga) : "memory");
      bun.q = *reinterpret_cast<const uint4*>(Wgp + (size_t)nk * N);
    }

    // A fragment: lane holds row (wv*16+ln), K = {8*half+0..7, 16+8*half+0..7}
    union FragA { v16bf v; uint4 q[2]; } af;
    const uint4* pa =
        reinterpret_cast<const uint4*>(&As[buf][(wv * 16 + ln) * KT]);
    af.q[0] = pa[half];
    af.q[1] = pa[2 + half];

    // B fragments (all four preloaded so WMMAs issue back-to-back):
    // lane holds col (j*16+ln), K = {8*half+0..7, 16+8*half+0..7}
    union FragB { v16bf v; uint4 q[2]; } bf[4];
    const uint4* pb = reinterpret_cast<const uint4*>(&Bs[buf][0]);
#pragma unroll
    for (int j = 0; j < 4; ++j) {
      bf[j].q[0] = pb[(j * 16 + ln) * 4 + half];
      bf[j].q[1] = pb[(j * 16 + ln) * 4 + 2 + half];
    }
#pragma unroll
    for (int j = 0; j < 4; ++j)
      acc[j] = __builtin_amdgcn_wmma_f32_16x16x32_bf16(
          false, af.v, false, bf[j].v, (short)0, acc[j], false, false);

    if (nk < K) {
      // finish staging next buffer: transpose-scatter B, drain async A DMA
#pragma unroll
      for (int ii = 0; ii < 8; ++ii)
        Bs[buf ^ 1][(bn + ii) * KT + bk] = bun.s[ii];
      asm volatile("s_wait_asynccnt 0x0" ::: "memory");
      __syncthreads();
      buf ^= 1;
    }
  }

  // epilogue: C layout m = i + 8*half, n = lane%16
#pragma unroll
  for (int j = 0; j < 4; ++j) {
    int ng = colbase + j * 16 + ln;
    float bv = bias[ng];
#pragma unroll
    for (int i = 0; i < 8; ++i) {
      int mg = rowbase + wv * 16 + half * 8 + i;
      C[(size_t)mg * N + ng] = acc[j][i] + bv;
    }
  }
}

// ---------------- temporal attention over heads (per position) -------------
__global__ __launch_bounds__(256)
void k_temporal(const float* __restrict__ qf, const float* __restrict__ kf,
                const float* __restrict__ vf, float* __restrict__ t_out)
{
  int idx = blockIdx.x * 256 + threadIdx.x;      // < NN*NH
  int n = idx >> 3, h = idx & 7;
  const float* q = qf + (size_t)n * HID + h * DD;
  float sc[NH];
  float mx = -3.4e38f;
#pragma unroll
  for (int g = 0; g < NH; ++g) {
    const float* kk = kf + (size_t)n * HID + g * DD;
    float s = 0.f;
#pragma unroll
    for (int d = 0; d < DD; ++d) s += q[d] * kk[d];
    s *= SCALE;
    sc[g] = s;
    mx = fmaxf(mx, s);
  }
  float sum = 0.f;
#pragma unroll
  for (int g = 0; g < NH; ++g) { sc[g] = __expf(sc[g] - mx); sum += sc[g]; }
  float inv = 1.f / sum;
  float outv[DD];
#pragma unroll
  for (int d = 0; d < DD; ++d) outv[d] = 0.f;
#pragma unroll
  for (int g = 0; g < NH; ++g) {
    float w = sc[g] * inv;
    const float* vv = vf + (size_t)n * HID + g * DD;
#pragma unroll
    for (int d = 0; d < DD; ++d) outv[d] += w * vv[d];
  }
  float* o = t_out + (size_t)n * HID + h * DD;
#pragma unroll
  for (int d = 0; d < DD; ++d) o[d] = outv[d];
}

// ---------------- graph branch ----------------
// pass 1: per-edge-per-head score + per-block max (h preserved: strides >= 8)
__global__ __launch_bounds__(256)
void k_edge_score_max(const float* __restrict__ gq, const float* __restrict__ gk,
                      const long long* __restrict__ eidx,
                      float* __restrict__ scores, float* __restrict__ pmax)
{
  const int tid = threadIdx.x;
  int idx = blockIdx.x * 256 + tid;              // < BE*NH
  int be = idx >> 3, h = idx & 7;
  int b = be >> 15;                              // be / EE
  int e = be & (EE - 1);
  int row = b * SS + (int)eidx[e];
  int col = b * SS + (int)eidx[EE + e];
  const float* q = gq + (size_t)row * HID + h * DD;
  const float* k = gk + (size_t)col * HID + h * DD;
  float s = 0.f;
#pragma unroll
  for (int d = 0; d < DD; ++d) s += q[d] * k[d];
  s *= SCALE;
  scores[idx] = s;

  __shared__ float red[256];
  red[tid] = s; __syncthreads();
  for (int st = 128; st >= 8; st >>= 1) {
    if (tid < st) red[tid] = fmaxf(red[tid], red[tid + st]);
    __syncthreads();
  }
  if (tid < 8) pmax[(size_t)blockIdx.x * 8 + tid] = red[tid];
}

// reduce per-head partials (max or sum), one block per head
__global__ __launch_bounds__(256)
void k_reduce_head(const float* __restrict__ part, int nblk,
                   float* __restrict__ out, int is_max)
{
  int h = blockIdx.x, tid = threadIdx.x;
  float acc = is_max ? -3.4e38f : 0.f;
  for (int i = tid; i < nblk; i += 256) {
    float v = part[(size_t)i * 8 + h];
    acc = is_max ? fmaxf(acc, v) : (acc + v);
  }
  __shared__ float red[256];
  red[tid] = acc; __syncthreads();
  for (int st = 128; st >= 1; st >>= 1) {
    if (tid < st)
      red[tid] = is_max ? fmaxf(red[tid], red[tid + st]) : (red[tid] + red[tid + st]);
    __syncthreads();
  }
  if (tid == 0) out[h] = red[0];
}

// pass 2: exp(score - max) in place + per-block sum
__global__ __launch_bounds__(256)
void k_edge_exp_sum(float* __restrict__ scores, const float* __restrict__ hmax,
                    float* __restrict__ psum)
{
  const int tid = threadIdx.x;
  int idx = blockIdx.x * 256 + tid;
  int h = idx & 7;
  float w = __expf(scores[idx] - hmax[h]);
  scores[idx] = w;
  __shared__ float red[256];
  red[tid] = w; __syncthreads();
  for (int st = 128; st >= 8; st >>= 1) {
    if (tid < st) red[tid] += red[tid + st];
    __syncthreads();
  }
  if (tid < 8) psum[(size_t)blockIdx.x * 8 + tid] = red[tid];
}

// pass 3: scatter-add weighted v_j into agg[row]
__global__ __launch_bounds__(256)
void k_edge_scatter(const float* __restrict__ scores, const float* __restrict__ hsum,
                    const float* __restrict__ gv, const long long* __restrict__ eidx,
                    float* __restrict__ agg)
{
  int idx = blockIdx.x * 256 + threadIdx.x;
  int be = idx >> 3, h = idx & 7;
  int b = be >> 15;
  int e = be & (EE - 1);
  int row = b * SS + (int)eidx[e];
  int col = b * SS + (int)eidx[EE + e];
  float w = scores[idx] / hsum[h];
  const float* v = gv + (size_t)col * HID + h * DD;
  float* dst = agg + (size_t)row * HID + h * DD;
#pragma unroll
  for (int d = 0; d < DD; ++d) atomicAdd(&dst[d], w * v[d]);
}

// ---------------- fusion ----------------
__global__ void k_build_cat(const float* __restrict__ t_out,
                            const float* __restrict__ g_out,
                            unsigned short* __restrict__ cat, int total)
{
  int i = blockIdx.x * 256 + threadIdx.x;
  if (i >= total) return;
  int n = i >> 8, c = i & 255;
  cat[(size_t)n * (2 * HID) + c]       = f2bf(t_out[i]);
  cat[(size_t)n * (2 * HID) + HID + c] = f2bf(g_out[i]);
}

__global__ void k_gate_fuse(const float* __restrict__ logits,
                            const float* __restrict__ t_out,
                            const float* __restrict__ g_out,
                            unsigned short* __restrict__ fusedb, int n)
{
  int i = blockIdx.x * 256 + threadIdx.x;
  if (i >= n) return;
  float f = 1.f / (1.f + __expf(-logits[i]));
  fusedb[i] = f2bf(f * t_out[i] + (1.f - f) * g_out[i]);
}

// ---------------- driver ----------------
extern "C" void kernel_launch(void* const* d_in, const int* in_sizes, int n_in,
                              void* d_out, int out_size, void* d_ws, size_t ws_size,
                              hipStream_t stream)
{
  (void)in_sizes; (void)n_in; (void)out_size; (void)ws_size;
  const float*     x    = (const float*)d_in[0];
  const long long* eidx = (const long long*)d_in[1];
  const float* Wtq = (const float*)d_in[2];  const float* btq = (const float*)d_in[3];
  const float* Wtk = (const float*)d_in[4];  const float* btk = (const float*)d_in[5];
  const float* Wtv = (const float*)d_in[6];  const float* btv = (const float*)d_in[7];
  const float* Wgq = (const float*)d_in[8];  const float* bgq = (const float*)d_in[9];
  const float* Wgk = (const float*)d_in[10]; const float* bgk = (const float*)d_in[11];
  const float* Wgv = (const float*)d_in[12]; const float* bgv = (const float*)d_in[13];
  const float* Wf  = (const float*)d_in[14]; const float* bfb = (const float*)d_in[15];
  const float* Wo  = (const float*)d_in[16]; const float* bo  = (const float*)d_in[17];
  float* out = (float*)d_out;

  char* ws = (char*)d_ws;
  size_t cur = 0;
  auto alloc = [&](size_t bytes) -> char* {
    char* p = ws + cur;
    cur += (bytes + 255) & ~(size_t)255;
    return p;
  };

  unsigned short* xb     = (unsigned short*)alloc((size_t)NN * HID * 2);
  float*          qf     = (float*)alloc((size_t)NN * HID * 4);
  float*          kf     = (float*)alloc((size_t)NN * HID * 4);
  float*          vf     = (float*)alloc((size_t)NN * HID * 4);
  float*          t_out  = (float*)alloc((size_t)NN * HID * 4);
  float*          agg    = (float*)alloc((size_t)NN * HID * 4);
  float*          scores = (float*)alloc((size_t)BE * NH * 4);
  float*          pmax   = (float*)alloc((size_t)8192 * 8 * 4);
  float*          psum   = (float*)alloc((size_t)8192 * 8 * 4);
  float*          hmax   = (float*)alloc(256);
  float*          hsum   = (float*)alloc(256);
  unsigned short* catb   = (unsigned short*)alloc((size_t)NN * 2 * HID * 2);
  float*          glog   = (float*)alloc((size_t)NN * HID * 4);
  unsigned short* fusedb = (unsigned short*)alloc((size_t)NN * HID * 2);
  unsigned short* wb0    = (unsigned short*)alloc(65536 * 2);
  unsigned short* wb1    = (unsigned short*)alloc(65536 * 2);
  unsigned short* wb2    = (unsigned short*)alloc(65536 * 2);
  unsigned short* wbf    = (unsigned short*)alloc(131072 * 2);
  unsigned short* wbo    = (unsigned short*)alloc(65536 * 2);

  const int NE = NN * HID;                       // 4,194,304 elements
  dim3 blk(256);
  dim3 gCvtX((NE + 255) / 256);
  dim3 gCvtW((65536 + 255) / 256);
  dim3 gGemm(HID / TN, NN / TM);                 // (4, 128)
  dim3 gEdge(BE * NH / 256);                     // 8192

  // ---- precision demotion for WMMA ----
  k_f32_to_bf16<<<gCvtX, blk, 0, stream>>>(x, xb, NE);

  // ---- temporal branch ----
  k_f32_to_bf16<<<gCvtW, blk, 0, stream>>>(Wtq, wb0, 65536);
  k_f32_to_bf16<<<gCvtW, blk, 0, stream>>>(Wtk, wb1, 65536);
  k_f32_to_bf16<<<gCvtW, blk, 0, stream>>>(Wtv, wb2, 65536);
  k_gemm_bf16_wmma<<<gGemm, blk, 0, stream>>>(xb, wb0, btq, qf, NN, HID, HID);
  k_gemm_bf16_wmma<<<gGemm, blk, 0, stream>>>(xb, wb1, btk, kf, NN, HID, HID);
  k_gemm_bf16_wmma<<<gGemm, blk, 0, stream>>>(xb, wb2, btv, vf, NN, HID, HID);
  k_temporal<<<NN * NH / 256, blk, 0, stream>>>(qf, kf, vf, t_out);

  // ---- graph branch (reuse qf/kf/vf) ----
  k_f32_to_bf16<<<gCvtW, blk, 0, stream>>>(Wgq, wb0, 65536);
  k_f32_to_bf16<<<gCvtW, blk, 0, stream>>>(Wgk, wb1, 65536);
  k_f32_to_bf16<<<gCvtW, blk, 0, stream>>>(Wgv, wb2, 65536);
  k_gemm_bf16_wmma<<<gGemm, blk, 0, stream>>>(xb, wb0, bgq, qf, NN, HID, HID);
  k_gemm_bf16_wmma<<<gGemm, blk, 0, stream>>>(xb, wb1, bgk, kf, NN, HID, HID);
  k_gemm_bf16_wmma<<<gGemm, blk, 0, stream>>>(xb, wb2, bgv, vf, NN, HID, HID);

  k_zero_f32<<<gCvtX, blk, 0, stream>>>(agg, NE);
  k_edge_score_max<<<gEdge, blk, 0, stream>>>(qf, kf, eidx, scores, pmax);
  k_reduce_head<<<8, blk, 0, stream>>>(pmax, 8192, hmax, 1);
  k_edge_exp_sum<<<gEdge, blk, 0, stream>>>(scores, hmax, psum);
  k_reduce_head<<<8, blk, 0, stream>>>(psum, 8192, hsum, 0);
  k_edge_scatter<<<gEdge, blk, 0, stream>>>(scores, hsum, vf, eidx, agg);

  // ---- gated fusion + output projection ----
  k_build_cat<<<gCvtX, blk, 0, stream>>>(t_out, agg, catb, NE);
  k_f32_to_bf16<<<(131072 + 255) / 256, blk, 0, stream>>>(Wf, wbf, 131072);
  k_gemm_bf16_wmma<<<gGemm, blk, 0, stream>>>(catb, wbf, bfb, glog, NN, HID, 2 * HID);
  k_gate_fuse<<<gCvtX, blk, 0, stream>>>(glog, t_out, agg, fusedb, NE);
  k_f32_to_bf16<<<gCvtW, blk, 0, stream>>>(Wo, wbo, 65536);
  k_gemm_bf16_wmma<<<gGemm, blk, 0, stream>>>(fusedb, wbo, bo, out, NN, HID, HID);
}
